// Interaction_31490700214650
// MI455X (gfx1250) — matrix operand order repaired and checked by reading, not compiled, not run on hardware
//
#include <hip/hip_runtime.h>
#include <hip/hip_bf16.h>
#include <stdint.h>

// Problem constants (reference: B=4, N=2048, D=256)
#define BB 4
#define NN 2048
#define DD 256
#define NEGV  (-1000000000.0f)
#define ALPHA 0.2f

typedef __attribute__((ext_vector_type(16))) __bf16 v16bf;
typedef __attribute__((ext_vector_type(8)))  float  v8f;

// f32 -> bf16 bits (round to nearest even)
__device__ __forceinline__ uint32_t bfbits(float x) {
    union { float f; uint32_t u; } v; v.f = x;
    return (v.u + 0x7FFFu + ((v.u >> 16) & 1u)) >> 16;
}
__device__ __forceinline__ uint32_t pk2(float a, float b) {
    return bfbits(a) | (bfbits(b) << 16);
}
__device__ __forceinline__ __bf16 bfh(float x) {
    uint16_t h = (uint16_t)bfbits(x);
    __bf16 o; __builtin_memcpy(&o, &h, 2); return o;
}
// leaky_relu(x, 0.2) == max(x, 0.2*x)
__device__ __forceinline__ float leaky(float x) { return fmaxf(x, ALPHA * x); }

__device__ __forceinline__ float wave_sum(float v) {
#pragma unroll
    for (int off = 16; off > 0; off >>= 1) v += __shfl_xor(v, off, 32);
    return v;
}

__device__ __forceinline__ v8f wmma_bf16(v16bf a, v16bf b, v8f c) {
    return __builtin_amdgcn_wmma_f32_16x16x32_bf16(false, a, false, b, (short)0, c, false, false);
}

// ---------------------------------------------------------------------------
// One-time f32 row-major [K=256][NC=256] -> bf16 col-major [NC][K] weight
// conversion (weights live in L2; done once per weight instead of per tile).
// ---------------------------------------------------------------------------
__global__ __launch_bounds__(256) void convert_w(
    const float* __restrict__ Wt, __bf16* __restrict__ WtT)
{
    __shared__ float tile[16][17];
    int k0 = blockIdx.x * 16, c0 = blockIdx.y * 16;
    int a = threadIdx.x >> 4, t2 = threadIdx.x & 15;
    tile[a][t2] = Wt[(size_t)(k0 + a) * DD + c0 + t2];
    __syncthreads();
    WtT[(size_t)(c0 + a) * DD + k0 + t2] = bfh(tile[t2][a]);
}

// h (B,N,D) f32 -> hT (B,D,N) bf16 (transposed+converted once for attn P@V)
__global__ __launch_bounds__(256) void transpose_h(
    const float* __restrict__ h, __bf16* __restrict__ hT)
{
    __shared__ float tile[16][17];
    int b = blockIdx.z, j0 = blockIdx.x * 16, c0 = blockIdx.y * 16;
    int a = threadIdx.x >> 4, t2 = threadIdx.x & 15;
    tile[a][t2] = h[((size_t)b * NN + j0 + a) * DD + c0 + t2];
    __syncthreads();
    hT[((size_t)b * DD + c0 + a) * NN + j0 + t2] = bfh(tile[t2][a]);
}

// ---------------------------------------------------------------------------
// C[M x NC] = (A (+A2)) [M x K] @ W [K x NC] + bscale*bias.
// WtT is bf16 col-major [NC][K] (pre-converted): B operand = direct contiguous
// 32B global load per lane (no LDS). A staged via LDS: one float4 global load
// + one packed ds_store_b64 per thread per k-step.
// Block = 128 (4 waves), WG tile 16 rows x 64 cols, one wmma per wave per step.
// ---------------------------------------------------------------------------
template <bool HASA2>
__global__ __launch_bounds__(128) void gemm_xw(
    const float* __restrict__ A, const float* __restrict__ A2,
    const __bf16* __restrict__ WtT, const float* __restrict__ bias,
    float bscale, float* __restrict__ C, int M, int K, int NC)
{
    __shared__ uint64_t lA64[16 * 8];                 // 16x32 bf16 A tile
    const __bf16* lA = (const __bf16*)lA64;
    const int tid = threadIdx.x, wave = tid >> 5, lane = tid & 31;
    const int m0 = blockIdx.x * 16, n0 = blockIdx.y * 64;
    const int mr = tid >> 3, kk4 = (tid & 7) * 4;     // A staging map (float4/thread)
    const int mA = lane & 15, kA = (lane < 16) ? 0 : 16;
    const int nB = lane & 15, kB = (lane < 16) ? 0 : 16;
    const int col = n0 + wave * 16 + nB;
    const __bf16* bp = WtT + (size_t)col * K;
    v8f acc = {};
    for (int k0 = 0; k0 < K; k0 += 32) {
        size_t off = (size_t)(m0 + mr) * K + k0 + kk4;
        float4 av = *(const float4*)(A + off);
        if (HASA2) {
            float4 a2 = *(const float4*)(A2 + off);
            av.x += a2.x; av.y += a2.y; av.z += a2.z; av.w += a2.w;
        }
        lA64[mr * 8 + (kk4 >> 2)] =
            (uint64_t)pk2(av.x, av.y) | ((uint64_t)pk2(av.z, av.w) << 32);
        __syncthreads();
        v16bf a;
#pragma unroll
        for (int t = 0; t < 8; ++t) {
            a[t]     = lA[mA * 32 + kA + t];
            a[t + 8] = lA[mA * 32 + kA + 8 + t];
        }
        const v16bf b = *(const v16bf*)(bp + k0 + kB);
        acc = wmma_bf16(a, b, acc);
        __syncthreads();
    }
    const int rb = (lane < 16) ? 0 : 8;
    const float bb = bias ? bias[col] * bscale : 0.0f;
#pragma unroll
    for (int r = 0; r < 8; ++r)
        C[(size_t)(m0 + rb + r) * NC + col] = acc[r] + bb;
}

// ---------------------------------------------------------------------------
// s1 = h@a1 ; s2 = (h+evt)@a2     (one wave per row)
// ---------------------------------------------------------------------------
__global__ __launch_bounds__(256) void s1s2_kernel(
    const float* __restrict__ h, const float* __restrict__ evt,
    const float* __restrict__ a, float* __restrict__ s1, float* __restrict__ s2)
{
    int wave = threadIdx.x >> 5, lane = threadIdx.x & 31;
    int row = blockIdx.x * 8 + wave;
    float a1 = 0.f, a2 = 0.f;
    for (int d = lane; d < DD; d += 32) {
        float hv = h[(size_t)row * DD + d];
        a1 += hv * a[d];
        a2 += (hv + evt[(size_t)row * DD + d]) * a[DD + d];
    }
    a1 = wave_sum(a1); a2 = wave_sum(a2);
    if (lane == 0) { s1[row] = a1; s2[row] = a2; }
}

// ---------------------------------------------------------------------------
// Attention row stats: online (max, sum-exp) of leaky(s1[i]+s2[j])*m over j.
// Branchless; stores rmax and the RECIPROCAL of the sum (one divide per row,
// so the P@V epilogue is a pure multiply).
// ---------------------------------------------------------------------------
__global__ __launch_bounds__(256) void row_stats(
    const float* __restrict__ s1, const float* __restrict__ s2,
    const int* __restrict__ im, const int* __restrict__ om,
    float* __restrict__ rmax, float* __restrict__ rsuminv)
{
    int wave = threadIdx.x >> 5, lane = threadIdx.x & 31;
    int row = blockIdx.x * 8 + wave;           // row = b*N + i
    int b = row >> 11;
    float s1i = s1[row];
    float mx = -3.0e38f, sm = 0.0f;
    for (int j = lane; j < NN; j += 32) {
        size_t mo = (size_t)row * NN + j;
        int mm = im[mo] + om[mo];
        float x = leaky(s1i + s2[b * NN + j]) * (float)mm;
        float v = (mm == 0) ? NEGV : x;
        float nm = fmaxf(mx, v);
        sm = sm * __expf(mx - nm) + __expf(v - nm);
        mx = nm;
    }
#pragma unroll
    for (int off = 16; off > 0; off >>= 1) {
        float m2 = __shfl_xor(mx, off, 32);
        float s2v = __shfl_xor(sm, off, 32);
        float nm = fmaxf(mx, m2);
        sm = sm * __expf(mx - nm) + s2v * __expf(m2 - nm);
        mx = nm;
    }
    if (lane == 0) { rmax[row] = mx; rsuminv[row] = 1.0f / sm; }
}

// ---------------------------------------------------------------------------
// h_prime = softmax(e) @ h. P tiles (16x32) regenerated in-register from
// s1/s2/masks (rank-1 score structure), packed to LDS; h streamed as bf16
// directly from pre-transposed hT (contiguous 32B per lane). 4 wmma per step.
// ---------------------------------------------------------------------------
__global__ __launch_bounds__(128) void attn_pv(
    const __bf16* __restrict__ hT, const float* __restrict__ s1, const float* __restrict__ s2,
    const int* __restrict__ im, const int* __restrict__ om,
    const float* __restrict__ rmax, const float* __restrict__ rsuminv,
    float* __restrict__ hp)
{
    __shared__ uint64_t lA64[16 * 8];          // 16x32 bf16 P tile
    const __bf16* lA = (const __bf16*)lA64;
    const int tid = threadIdx.x, wave = tid >> 5, lane = tid & 31;
    const int b = blockIdx.y, i0 = blockIdx.x * 16;
    const int mr = tid >> 3, jj0 = (tid & 7) * 4;   // P staging map (4 j / thread)
    const float s1i = s1[b * NN + i0 + mr];
    const float rmx = rmax[b * NN + i0 + mr];
    const int mA = lane & 15, kA = (lane < 16) ? 0 : 16;
    const int nB = lane & 15, kB = (lane < 16) ? 0 : 16;
    v8f acc[4] = {};
    for (int j0 = 0; j0 < NN; j0 += 32) {
        size_t mo = ((size_t)b * NN + i0 + mr) * NN + j0 + jj0;
        int4 imv = *(const int4*)(im + mo);
        int4 omv = *(const int4*)(om + mo);
        float4 s2v = *(const float4*)(s2 + b * NN + j0 + jj0);
        const int* ip = (const int*)&imv;
        const int* op = (const int*)&omv;
        const float* sp = (const float*)&s2v;
        float p[4];
#pragma unroll
        for (int u = 0; u < 4; ++u) {
            int mm = ip[u] + op[u];
            float x = leaky(s1i + sp[u]) * (float)mm;
            float pv = __expf(x - rmx);              // unnormalized
            p[u] = (mm == 0) ? 0.0f : pv;
        }
        lA64[mr * 8 + (jj0 >> 2)] =
            (uint64_t)pk2(p[0], p[1]) | ((uint64_t)pk2(p[2], p[3]) << 32);
        __syncthreads();
        v16bf a;
#pragma unroll
        for (int t = 0; t < 8; ++t) { a[t] = lA[mA * 32 + kA + t]; a[t + 8] = lA[mA * 32 + kA + 8 + t]; }
#pragma unroll
        for (int ch = 0; ch < 4; ++ch) {
            const int col = wave * 64 + ch * 16 + nB;
            const v16bf bv = *(const v16bf*)(hT + ((size_t)b * DD + col) * NN + j0 + kB);
            acc[ch] = wmma_bf16(a, bv, acc[ch]);
        }
        __syncthreads();
    }
    const int rb = (lane < 16) ? 0 : 8;
    float inv[8];
#pragma unroll
    for (int r = 0; r < 8; ++r) inv[r] = rsuminv[b * NN + i0 + rb + r];
#pragma unroll
    for (int ch = 0; ch < 4; ++ch) {
        int col = wave * 64 + ch * 16 + nB;
#pragma unroll
        for (int r = 0; r < 8; ++r)
            hp[((size_t)b * NN + i0 + rb + r) * DD + col] = acc[ch][r] * inv[r];
    }
}

// ---------------------------------------------------------------------------
// Per-batch q = (tv[+temo])@qW + qb ; kq = kW@q ; wq = wW@q ; c0 = q·(kb+wb)
// ---------------------------------------------------------------------------
__global__ __launch_bounds__(256) void csk_prepare(
    const float* __restrict__ inp, const float* __restrict__ emo, const int* __restrict__ cl,
    const float* __restrict__ linW, const float* __restrict__ linb,
    float* __restrict__ q_out, float* __restrict__ kq_out, float* __restrict__ wq_out,
    float* __restrict__ c0_out)
{
    __shared__ float tvec[DD], qv[DD], red[256];
    int b = blockIdx.x, d = threadIdx.x;
    int t = cl[b] - 1;
    size_t o = ((size_t)b * NN + t) * DD + d;
    tvec[d] = inp[o] + (emo ? emo[o] : 0.0f);
    __syncthreads();
    const float* qW = linW + DD * DD;     // lin order: wW,qW,kW,vW
    const float* qb = linb + DD;
    float acc = qb[d];
    for (int c = 0; c < DD; ++c) acc += tvec[c] * qW[(size_t)c * DD + d];
    qv[d] = acc; q_out[b * DD + d] = acc;
    __syncthreads();
    const float* kW = linW + 2 * DD * DD;
    const float* wW = linW;
    float a1 = 0.f, a2 = 0.f;
    for (int dd = 0; dd < DD; ++dd) {
        float qq = qv[dd];
        a1 += kW[(size_t)d * DD + dd] * qq;
        a2 += wW[(size_t)d * DD + dd] * qq;
    }
    kq_out[b * DD + d] = a1; wq_out[b * DD + d] = a2;
    const float* kb = linb + 2 * DD;
    const float* wb = linb;
    red[d] = qv[d] * (kb[d] + wb[d]);
    __syncthreads();
    for (int s = 128; s > 0; s >>= 1) { if (d < s) red[d] += red[d + s]; __syncthreads(); }
    if (d == 0) c0_out[b] = red[0];
}

// ---------------------------------------------------------------------------
// scores[n] = inv*( ti*(hp·kq + ic·wq + c0) + to*(hp·kq + oc·wq + c0) )
// ---------------------------------------------------------------------------
__global__ __launch_bounds__(256) void csk_scores(
    const float* __restrict__ hp, const float* __restrict__ ic, const float* __restrict__ oc,
    const int* __restrict__ im, const int* __restrict__ om, const int* __restrict__ cl,
    const float* __restrict__ kq, const float* __restrict__ wq, const float* __restrict__ c0,
    float* __restrict__ scores)
{
    int wave = threadIdx.x >> 5, lane = threadIdx.x & 31;
    int row = blockIdx.x * 8 + wave;
    int b = row >> 11, n = row & (NN - 1);
    int t = cl[b] - 1;
    size_t mo = ((size_t)b * NN + t) * NN + n;
    float ti = (float)im[mo], to = (float)om[mo];
    float dh = 0.f, di = 0.f, dq = 0.f;
    for (int d = lane; d < DD; d += 32) {
        size_t o = (size_t)row * DD + d;
        dh += hp[o] * kq[b * DD + d];
        di += ic[o] * wq[b * DD + d];
        dq += oc[o] * wq[b * DD + d];
    }
    dh = wave_sum(dh); di = wave_sum(di); dq = wave_sum(dq);
    if (lane == 0) {
        float base = dh + c0[b];
        float si = ti * (base + di), so = to * (base + dq);
        float mm = ti + to;
        scores[row] = (mm == 0.0f) ? NEGV : 0.0625f * (si + so);   // inv = 1/sqrt(256)
    }
}

// ---------------------------------------------------------------------------
__global__ __launch_bounds__(256) void softmax_row(
    const float* __restrict__ sc, float* __restrict__ pr)
{
    __shared__ float red[256];
    int b = blockIdx.x, t = threadIdx.x;
    float mx = -3.0e38f;
    for (int j = t; j < NN; j += 256) mx = fmaxf(mx, sc[b * NN + j]);
    red[t] = mx; __syncthreads();
    for (int s = 128; s > 0; s >>= 1) { if (t < s) red[t] = fmaxf(red[t], red[t + s]); __syncthreads(); }
    mx = red[0]; __syncthreads();
    float sm = 0.f;
    for (int j = t; j < NN; j += 256) sm += __expf(sc[b * NN + j] - mx);
    red[t] = sm; __syncthreads();
    for (int s = 128; s > 0; s >>= 1) { if (t < s) red[t] += red[t + s]; __syncthreads(); }
    float is = 1.0f / red[0];
    for (int j = t; j < NN; j += 256) pr[b * NN + j] = __expf(sc[b * NN + j] - mx) * is;
}

// ---------------------------------------------------------------------------
// out = relu( hp + inp + conn_e + conn_a )
// ---------------------------------------------------------------------------
__global__ __launch_bounds__(256) void final_out(
    const float* __restrict__ inp, const float* __restrict__ hp,
    const float* __restrict__ vbe, const float* __restrict__ wie, const float* __restrict__ woe,
    const float* __restrict__ vba, const float* __restrict__ wia, const float* __restrict__ woa,
    const float* __restrict__ qe, const float* __restrict__ qa,
    const float* __restrict__ pe, const float* __restrict__ pa,
    const int* __restrict__ im, const int* __restrict__ om, const int* __restrict__ cl,
    float* __restrict__ out)
{
    int row = blockIdx.x; int b = row >> 11, n = row & (NN - 1);
    int d = threadIdx.x;
    int t = cl[b] - 1;
    size_t mo = ((size_t)b * NN + t) * NN + n;
    float ti = (float)im[mo], to = (float)om[mo];
    float Pe = pe[row], Pa = pa[row];
    size_t o = (size_t)row * DD + d;
    float ce = Pe * ((ti + to) * qe[b * DD + d] + ti * (vbe[o] + wie[o]) + to * (vbe[o] + woe[o]));
    float ca = Pa * ((ti + to) * qa[b * DD + d] + ti * (vba[o] + wia[o]) + to * (vba[o] + woa[o]));
    float v = hp[o] + inp[o] + ce + ca;
    out[o] = fmaxf(v, 0.0f);
}

// ---------------------------------------------------------------------------
extern "C" void kernel_launch(void* const* d_in, const int* in_sizes, int n_in,
                              void* d_out, int out_size, void* d_ws, size_t ws_size,
                              hipStream_t stream)
{
    (void)in_sizes; (void)n_in; (void)out_size; (void)ws_size;
    const float* inp  = (const float*)d_in[0];
    const float* emo  = (const float*)d_in[1];
    const float* evb  = (const float*)d_in[2];
    const float* eva  = (const float*)d_in[3];
    const float* eic  = (const float*)d_in[4];
    const float* eoc  = (const float*)d_in[5];
    const float* iic  = (const float*)d_in[6];
    const float* ioc  = (const float*)d_in[7];
    const float* W    = (const float*)d_in[8];
    const float* avec = (const float*)d_in[9];
    const float* cskW = (const float*)d_in[10];
    const float* cskb = (const float*)d_in[11];
    const float* eW   = (const float*)d_in[12];
    const float* eb   = (const float*)d_in[13];
    const float* aW   = (const float*)d_in[14];
    const float* ab   = (const float*)d_in[15];
    const int*   cl   = (const int*)d_in[16];
    const int*   im   = (const int*)d_in[17];
    const int*   om   = (const int*)d_in[18];

    float* ws = (float*)d_ws;
    const size_t BIG = (size_t)BB * NN * DD;
    float* h   = ws;
    float* evt = ws + 1 * BIG;
    float* hp  = ws + 2 * BIG;
    float* vbe = ws + 3 * BIG;
    float* wie = ws + 4 * BIG;
    float* woe = ws + 5 * BIG;
    float* vba = ws + 6 * BIG;
    float* wia = ws + 7 * BIG;
    float* woa = ws + 8 * BIG;
    float* p   = ws + 9 * BIG;
    float* s1   = p;               p += (size_t)BB * NN;
    float* s2   = p;               p += (size_t)BB * NN;
    float* rmx  = p;               p += (size_t)BB * NN;
    float* rsm  = p;               p += (size_t)BB * NN;
    float* sce  = p;               p += (size_t)BB * NN;
    float* sca  = p;               p += (size_t)BB * NN;
    float* qe   = p;               p += (size_t)BB * DD;
    float* qa   = p;               p += (size_t)BB * DD;
    float* kqe  = p;               p += (size_t)BB * DD;
    float* wqe  = p;               p += (size_t)BB * DD;
    float* kqa  = p;               p += (size_t)BB * DD;
    float* wqa  = p;               p += (size_t)BB * DD;
    float* c0e  = p;               p += BB;
    float* c0a  = p;

    // bf16 scratch region (32B-aligned: 10*BIG floats from 256B-aligned base)
    __bf16* bfb   = (__bf16*)(ws + 10 * BIG);
    const size_t WSZ = (size_t)DD * DD;
    __bf16* wbfW  = bfb;                 // W
    __bf16* wbfC  = bfb + 1 * WSZ;       // csk_W
    __bf16* wbfVe = bfb + 2 * WSZ;       // emo vW
    __bf16* wbfWe = bfb + 3 * WSZ;       // emo wW
    __bf16* wbfVa = bfb + 4 * WSZ;       // act vW
    __bf16* wbfWa = bfb + 5 * WSZ;       // act wW
    __bf16* hT    = bfb + 6 * WSZ;       // h transposed bf16 (B,D,N)

    float* out = (float*)d_out;
    float* pe  = out + BIG;                 // emo_probs (B,1,N)
    float* pa  = pe + (size_t)BB * NN;      // act_probs (B,1,N)

    const dim3 gcv(DD / 16, DD / 16);
    convert_w<<<gcv, 256, 0, stream>>>(W,                wbfW);
    convert_w<<<gcv, 256, 0, stream>>>(cskW,             wbfC);
    convert_w<<<gcv, 256, 0, stream>>>(eW + 3 * DD * DD, wbfVe);
    convert_w<<<gcv, 256, 0, stream>>>(eW,               wbfWe);
    convert_w<<<gcv, 256, 0, stream>>>(aW + 3 * DD * DD, wbfVa);
    convert_w<<<gcv, 256, 0, stream>>>(aW,               wbfWa);

    const dim3 gw(BB * NN / 16, DD / 64);
    gemm_xw<false><<<gw, 128, 0, stream>>>(inp, nullptr, wbfW, nullptr, 0.f, h,   BB * NN, DD, DD);
    gemm_xw<true ><<<gw, 128, 0, stream>>>(evb, eva,     wbfC, cskb,    2.f, evt, BB * NN, DD, DD);
    s1s2_kernel<<<BB * NN / 8, 256, 0, stream>>>(h, evt, avec, s1, s2);
    row_stats  <<<BB * NN / 8, 256, 0, stream>>>(s1, s2, im, om, rmx, rsm);
    transpose_h<<<dim3(NN / 16, DD / 16, BB), 256, 0, stream>>>(h, hT);
    attn_pv<<<dim3(NN / 16, BB), 128, 0, stream>>>(hT, s1, s2, im, om, rmx, rsm, hp);
    // csk projection GEMMs (lin order: wW,qW,kW,vW)
    gemm_xw<false><<<gw, 128, 0, stream>>>(hp,  nullptr, wbfVe, eb + 3 * DD, 1.f, vbe, BB * NN, DD, DD);
    gemm_xw<false><<<gw, 128, 0, stream>>>(eic, nullptr, wbfWe, eb,          1.f, wie, BB * NN, DD, DD);
    gemm_xw<false><<<gw, 128, 0, stream>>>(eoc, nullptr, wbfWe, eb,          1.f, woe, BB * NN, DD, DD);
    gemm_xw<false><<<gw, 128, 0, stream>>>(hp,  nullptr, wbfVa, ab + 3 * DD, 1.f, vba, BB * NN, DD, DD);
    gemm_xw<false><<<gw, 128, 0, stream>>>(iic, nullptr, wbfWa, ab,          1.f, wia, BB * NN, DD, DD);
    gemm_xw<false><<<gw, 128, 0, stream>>>(ioc, nullptr, wbfWa, ab,          1.f, woa, BB * NN, DD, DD);
    csk_prepare<<<BB, 256, 0, stream>>>(inp, emo,     cl, eW, eb, qe, kqe, wqe, c0e);
    csk_prepare<<<BB, 256, 0, stream>>>(inp, nullptr, cl, aW, ab, qa, kqa, wqa, c0a);
    csk_scores<<<BB * NN / 8, 256, 0, stream>>>(hp, eic, eoc, im, om, cl, kqe, wqe, c0e, sce);
    csk_scores<<<BB * NN / 8, 256, 0, stream>>>(hp, iic, ioc, im, om, cl, kqa, wqa, c0a, sca);
    softmax_row<<<BB, 256, 0, stream>>>(sce, pe);
    softmax_row<<<BB, 256, 0, stream>>>(sca, pa);
    final_out<<<BB * NN, 256, 0, stream>>>(inp, hp, vbe, wie, woe, vba, wia, woa,
                                           qe, qa, pe, pa, im, om, cl, out);
}
